// HierarchicalGoldPredictor_14851996910058
// MI455X (gfx1250) — compile-verified, matrix operand order checked by reading, not run on hardware
//
#include <hip/hip_runtime.h>

// ---------------------------------------------------------------------------
// Fully-fused HierarchicalGoldPredictor for MI455X (gfx1250, wave32, WMMA).
// One wave processes 16 samples end-to-end; every Linear layer is a
// v_wmma_f32_16x16x32_f16 tile GEMM with weights resident in LDS (f16),
// activations staged per-wave in LDS. BatchNorm folded to y = x*s + t.
// din<=16 layers use a half-K fragment path (1 ds_load_b128 per fragment,
// upper K zeroed in registers). Q/K/V read A-fragments directly from the
// player buffer via a per-lane row remap (no staging copies).
// ---------------------------------------------------------------------------

typedef __attribute__((ext_vector_type(16))) _Float16 v16h;
typedef __attribute__((ext_vector_type(8)))  _Float16 v8h;
typedef __attribute__((ext_vector_type(8)))  float    v8f;

#define WAIT_DS() asm volatile("s_wait_dscnt 0" ::: "memory")

// ----- input pointer indices (x1,x2,g1..g5 then params in jax tree order) --
enum {
  IN_X1 = 0, IN_X2, IN_G1, IN_G2, IN_G3, IN_G4, IN_G5,
  A0_B, A0_W, A1_BETA, A1_GAMMA, A1_RM, A1_RV, A2_B, A2_W, A3_B, A3_W,
  B0_B, B0_W, B1_BETA, B1_GAMMA, B1_RM, B1_RV, B2_B, B2_W, B3_B, B3_W,
  C0_B, C0_W, C1_BETA, C1_GAMMA, C1_RM, C1_RV, C2_B, C2_W, C3_B, C3_W,
  DK_B, DK_W, DO0_B, DO0_W, DO1_B, DO1_W, DQ_B, DQ_W, DV_B, DV_W,
  EK_B, EK_W, EO0_B, EO0_W, EO1_B, EO1_W, EQ_B, EQ_W, EV_B, EV_W,
  F0_B, F0_W, F1_BETA, F1_GAMMA, F1_RM, F1_RV, F2_B, F2_W, F3_B, F3_W,
  G10_B, G10_W, G11_B, G11_W, G12_B, G12_W,
  G20_B, G20_W, G21_B, G21_W,
  G30_B, G30_W, G31_B, G31_W,
  G40_B, G40_W, G41_B, G41_W,
  G50_B, G50_W, G51_B, G51_W,
  FIN0_B, FIN0_W, FIN1_B, FIN1_W,
  N_PTRS // == 93
};

struct P93 { const float* p[N_PTRS]; };

// ----- LDS weight region offsets (in halfs), layout [dout_pad][Kpad] -------
constexpr int WH_A0  = 0;               // 64x32
constexpr int WH_A2  = WH_A0  + 64*32;  // 32x64
constexpr int WH_B0  = WH_A2  + 32*64;  // 32x32
constexpr int WH_B2  = WH_B0  + 32*32;  // 16x32
constexpr int WH_C0  = WH_B2  + 16*32;  // 64x32
constexpr int WH_C2  = WH_C0  + 64*32;  // 32x64
constexpr int WH_F0  = WH_C2  + 32*64;  // 64x64
constexpr int WH_F2  = WH_F0  + 64*64;  // 32x64
constexpr int WH_F3  = WH_F2  + 32*64;  // 16x32
constexpr int WH_G10 = WH_F3  + 16*32;  // 64x32
constexpr int WH_G11 = WH_G10 + 64*32;  // 32x64
constexpr int WH_G12 = WH_G11 + 32*64;  // 16x32 (8 valid rows)
constexpr int WH_G20 = WH_G12 + 16*32;  // 32x32
constexpr int WH_G21 = WH_G20 + 32*32;  // 16x32
constexpr int WH_G30 = WH_G21 + 16*32;  // 32x32
constexpr int WH_G31 = WH_G30 + 32*32;  // 16x32
constexpr int WH_G40 = WH_G31 + 16*32;  // 16x32
constexpr int WH_G41 = WH_G40 + 16*32;  // 16x32 (din 16)
constexpr int WH_G50 = WH_G41 + 16*32;  // 16x32
constexpr int WH_G51 = WH_G50 + 16*32;  // 16x32 (din 16)
constexpr int WH_DQ  = WH_G51 + 16*32;  // 16x32 (din 16)
constexpr int WH_DK  = WH_DQ  + 16*32;
constexpr int WH_DV  = WH_DK  + 16*32;
constexpr int WH_DO0 = WH_DV  + 16*32;  // 32x32 (din 16)
constexpr int WH_DO1 = WH_DO0 + 32*32;  // 16x32
constexpr int WH_EQ  = WH_DO1 + 16*32;
constexpr int WH_EK  = WH_EQ  + 16*32;
constexpr int WH_EV  = WH_EK  + 16*32;
constexpr int WH_EO0 = WH_EV  + 16*32;  // 32x32 (din 16)
constexpr int WH_EO1 = WH_EO0 + 32*32;  // 16x32
constexpr int WH_END = WH_EO1 + 16*32;  // 32256 halfs = 64512 B

// ----- LDS f32 vector region offsets (in floats) ---------------------------
constexpr int VF_A0B  = 0;
constexpr int VF_AS   = VF_A0B  + 64;
constexpr int VF_AT   = VF_AS   + 64;
constexpr int VF_A2B  = VF_AT   + 64;
constexpr int VF_A3W  = VF_A2B  + 32;
constexpr int VF_A3B  = VF_A3W  + 32;
constexpr int VF_B0B  = VF_A3B  + 1;
constexpr int VF_BS   = VF_B0B  + 32;
constexpr int VF_BT   = VF_BS   + 32;
constexpr int VF_B2B  = VF_BT   + 32;
constexpr int VF_B3W  = VF_B2B  + 16;
constexpr int VF_B3B  = VF_B3W  + 16;
constexpr int VF_C0B  = VF_B3B  + 1;
constexpr int VF_CS   = VF_C0B  + 64;
constexpr int VF_CT   = VF_CS   + 64;
constexpr int VF_C2B  = VF_CT   + 64;
constexpr int VF_C3W  = VF_C2B  + 32;
constexpr int VF_C3B  = VF_C3W  + 32;
constexpr int VF_F0B  = VF_C3B  + 1;
constexpr int VF_FS   = VF_F0B  + 64;
constexpr int VF_FT   = VF_FS   + 64;
constexpr int VF_F2B  = VF_FT   + 64;
constexpr int VF_F3B  = VF_F2B  + 32;
constexpr int VF_G10B = VF_F3B  + 16;
constexpr int VF_G11B = VF_G10B + 64;
constexpr int VF_G12B = VF_G11B + 32;
constexpr int VF_G20B = VF_G12B + 8;
constexpr int VF_G21B = VF_G20B + 32;
constexpr int VF_G30B = VF_G21B + 8;
constexpr int VF_G31B = VF_G30B + 32;
constexpr int VF_G40B = VF_G31B + 8;
constexpr int VF_G41B = VF_G40B + 16;
constexpr int VF_G50B = VF_G41B + 8;
constexpr int VF_G51B = VF_G50B + 16;
constexpr int VF_DQB  = VF_G51B + 8;
constexpr int VF_DKB  = VF_DQB  + 16;
constexpr int VF_DVB  = VF_DKB  + 16;
constexpr int VF_DO0B = VF_DVB  + 16;
constexpr int VF_DO1B = VF_DO0B + 32;
constexpr int VF_EQB  = VF_DO1B + 16;
constexpr int VF_EKB  = VF_EQB  + 16;
constexpr int VF_EVB  = VF_EKB  + 16;
constexpr int VF_EO0B = VF_EVB  + 16;
constexpr int VF_EO1B = VF_EO0B + 32;
constexpr int VF_FIN0B= VF_EO1B + 16;
constexpr int VF_FIN0W= VF_FIN0B+ 16;   // 48 (16x3, VALU path)
constexpr int VF_FIN1W= VF_FIN0W+ 48;
constexpr int VF_FIN1B= VF_FIN1W+ 16;
constexpr int VF_END  = VF_FIN1B+ 1;

// ----- per-wave staging region (offsets from wave base) --------------------
constexpr int WO_ACT0_H = 0;      // [16][64] f16
constexpr int WO_ACT1_H = 1024;   // [16][64] f16
constexpr int WO_GC_H   = 2048;   // [16][64] f16
constexpr int WO_PLY_H  = 3072;   // [16][10][16] f16
constexpr int WO_QKV_H  = 5632;   // 3 x [80][16] f16 (rows = sample*5+player)
constexpr int WO_SCW_B  = 18944;  // [16][5][5] f32
constexpr int WO_SC3_B  = 20544;  // [16][3] f32
constexpr int WAVE_BYTES = 20736;

constexpr int VEC_OFF  = WH_END * 2;
constexpr int WAVE_OFF = (VEC_OFF + VF_END * 4 + 63) & ~63;
constexpr int SMEM_BYTES = WAVE_OFF + 4 * WAVE_BYTES;   // ~154 KB < 320 KB WGP LDS

// ----- table-driven loaders (keeps one param pointer live at a time) -------
struct WDesc { int off; short idx, dout, din, rpad, kp; };
struct VDesc { int off; short idx, n; };

static __device__ const WDesc g_wtab[] = {
  {WH_A0,  A0_W, 64,20,64,32}, {WH_A2,  A2_W, 32,64,32,64},
  {WH_B0,  B0_W, 32,10,32,32}, {WH_B2,  B2_W, 16,32,16,32},
  {WH_C0,  C0_W, 64,32,64,32}, {WH_C2,  C2_W, 32,64,32,64},
  {WH_F0,  F0_W, 64,40,64,64}, {WH_F2,  F2_W, 32,64,32,64}, {WH_F3, F3_W, 16,32,16,32},
  {WH_G10, G10_W,64,12,64,32}, {WH_G11, G11_W,32,64,32,64}, {WH_G12,G12_W, 8,32,16,32},
  {WH_G20, G20_W,32, 8,32,32}, {WH_G21, G21_W, 8,32,16,32},
  {WH_G30, G30_W,32, 6,32,32}, {WH_G31, G31_W, 8,32,16,32},
  {WH_G40, G40_W,16, 6,16,32}, {WH_G41, G41_W, 8,16,16,32},
  {WH_G50, G50_W,16, 4,16,32}, {WH_G51, G51_W, 8,16,16,32},
  {WH_DQ,  DQ_W, 16,16,16,32}, {WH_DK,  DK_W, 16,16,16,32}, {WH_DV, DV_W, 16,16,16,32},
  {WH_DO0, DO0_W,32,16,32,32}, {WH_DO1, DO1_W,16,32,16,32},
  {WH_EQ,  EQ_W, 16,16,16,32}, {WH_EK,  EK_W, 16,16,16,32}, {WH_EV, EV_W, 16,16,16,32},
  {WH_EO0, EO0_W,32,16,32,32}, {WH_EO1, EO1_W,16,32,16,32},
};

static __device__ const VDesc g_vtab[] = {
  {VF_A0B, A0_B,64}, {VF_A2B, A2_B,32}, {VF_A3W, A3_W,32}, {VF_A3B, A3_B,1},
  {VF_B0B, B0_B,32}, {VF_B2B, B2_B,16}, {VF_B3W, B3_W,16}, {VF_B3B, B3_B,1},
  {VF_C0B, C0_B,64}, {VF_C2B, C2_B,32}, {VF_C3W, C3_W,32}, {VF_C3B, C3_B,1},
  {VF_F0B, F0_B,64}, {VF_F2B, F2_B,32}, {VF_F3B, F3_B,16},
  {VF_G10B,G10_B,64},{VF_G11B,G11_B,32},{VF_G12B,G12_B,8},
  {VF_G20B,G20_B,32},{VF_G21B,G21_B,8},
  {VF_G30B,G30_B,32},{VF_G31B,G31_B,8},
  {VF_G40B,G40_B,16},{VF_G41B,G41_B,8},
  {VF_G50B,G50_B,16},{VF_G51B,G51_B,8},
  {VF_DQB, DQ_B,16}, {VF_DKB, DK_B,16}, {VF_DVB, DV_B,16},
  {VF_DO0B,DO0_B,32},{VF_DO1B,DO1_B,16},
  {VF_EQB, EQ_B,16}, {VF_EKB, EK_B,16}, {VF_EVB, EV_B,16},
  {VF_EO0B,EO0_B,32},{VF_EO1B,EO1_B,16},
  {VF_FIN0B,FIN0_B,16},{VF_FIN0W,FIN0_W,48},{VF_FIN1W,FIN1_W,16},{VF_FIN1B,FIN1_B,1},
};

// ---------------------------------------------------------------------------
__device__ __forceinline__ v16h ldfrag(const _Float16* p) {
  // 16-bit A/B fragment: halves 0..7 = K kb..kb+7, halves 8..15 = K kb+16..kb+23
  v8h lo = *(const v8h*)(p);
  v8h hi = *(const v8h*)(p + 16);
  v16h r;
#pragma unroll
  for (int i = 0; i < 8; ++i) { r[i] = lo[i]; r[i + 8] = hi[i]; }
  return r;
}
__device__ __forceinline__ v16h ldfragK16(const _Float16* p) {
  // din<=16 fast path: one b128 load, upper-K halves are zero.
  v8h lo = *(const v8h*)(p);
  v16h r = {};
#pragma unroll
  for (int i = 0; i < 8; ++i) r[i] = lo[i];
  return r;
}

// out = act(in[16][kp] @ W[nout][kp]^T + bias) -> LDS f16. A-frags hoisted.
__device__ __forceinline__ void dense(const _Float16* in, int kp, const _Float16* W,
                                      const float* bias, const float* bnS, const float* bnT,
                                      bool lrelu, _Float16* out, int ostride,
                                      int nout, int noutPad, int lane) {
  const int nl = lane & 15;
  const int kb = (lane & 16) ? 8 : 0;
  const int mb = kb;
  const bool two = (kp > 32);
  v16h a0 = ldfrag(in + nl * kp + kb);
  v16h a1 = {};
  if (two) a1 = ldfrag(in + nl * kp + 32 + kb);
  for (int n0 = 0; n0 < noutPad; n0 += 16) {
    const int n = n0 + nl;
    v8f c = {};
    if (n0 < nout) {  // wave-uniform: EXEC all-ones around WMMA
      v16h b0 = ldfrag(W + n * kp + kb);
      c = __builtin_amdgcn_wmma_f32_16x16x32_f16(false, a0, false, b0, (short)0, c,
                                                 false, false);
      if (two) {
        v16h b1 = ldfrag(W + n * kp + 32 + kb);
        c = __builtin_amdgcn_wmma_f32_16x16x32_f16(false, a1, false, b1, (short)0, c,
                                                   false, false);
      }
    }
    const bool valid = (n < nout);
    const float bb = valid ? bias[n] : 0.f;
    const float s  = (bnS && valid) ? bnS[n] : 1.f;
    const float t  = (bnS && valid) ? bnT[n] : 0.f;
#pragma unroll
    for (int j = 0; j < 8; ++j) {
      float v = c[j] + bb;
      if (lrelu) v = (v > 0.f) ? v : 0.2f * v;
      v = v * s + t;
      if (!valid) v = 0.f;
      out[(mb + j) * ostride + n] = (_Float16)v;
    }
  }
  WAIT_DS();
}

// din<=16 layer: A row per lane at 'in + rowOffH' (16 contiguous halfs),
// W is [*][32] with K=16..31 zero. One b128 per fragment.
__device__ __forceinline__ void dense_k16(const _Float16* in, int rowOffH,
                                          const _Float16* W, const float* bias, bool lrelu,
                                          _Float16* out, int ostride, int nout, int noutPad,
                                          int lane) {
  const int nl = lane & 15;
  const int kb = (lane & 16) ? 8 : 0;
  const int mb = kb;
  v16h a = ldfragK16(in + rowOffH + kb);
  for (int n0 = 0; n0 < noutPad; n0 += 16) {
    const int n = n0 + nl;
    v8f c = {};
    if (n0 < nout) {
      v16h b = ldfragK16(W + n * 32 + kb);
      c = __builtin_amdgcn_wmma_f32_16x16x32_f16(false, a, false, b, (short)0, c,
                                                 false, false);
    }
    const bool valid = (n < nout);
    const float bb = valid ? bias[n] : 0.f;
#pragma unroll
    for (int j = 0; j < 8; ++j) {
      float v = c[j] + bb;
      if (lrelu) v = (v > 0.f) ? v : 0.2f * v;
      if (!valid) v = 0.f;
      out[(mb + j) * ostride + n] = (_Float16)v;
    }
  }
  WAIT_DS();
}

// Stage 16 samples x ncols f32 -> zero-padded f16 [16][32].
__device__ __forceinline__ void stage_in(const float* src, int sstride, int ncols,
                                         _Float16* dst, int lane) {
  for (int i = lane; i < 16 * 32; i += 32) {
    int m = i >> 5, k = i & 31;
    float v = (k < ncols) ? src[m * sstride + k] : 0.f;
    dst[i] = (_Float16)v;
  }
  WAIT_DS();
}

// ---------------------------------------------------------------------------
__global__ __launch_bounds__(128) void gold_kernel(P93 P, float* __restrict__ out) {
  extern __shared__ char smem[];
  _Float16* WH = (_Float16*)smem;
  float* V = (float*)(smem + VEC_OFF);
  const int tid = threadIdx.x;
  const int lane = tid & 31;
  const int wv = tid >> 5;

  // ---- cooperative weight load (table-driven; keeps SGPR pressure low) ----
#pragma clang loop unroll(disable)
  for (int j = 0; j < (int)(sizeof(g_wtab) / sizeof(g_wtab[0])); ++j) {
    const WDesc d = g_wtab[j];
    const float* src = P.p[d.idx];
    _Float16* dst = WH + d.off;
    const int tot = d.rpad * d.kp;
#pragma clang loop unroll(disable)
    for (int i = tid; i < tot; i += 128) {
      int r = i / d.kp, k = i - r * d.kp;
      float v = (r < d.dout && k < d.din) ? src[r * d.din + k] : 0.f;
      dst[i] = (_Float16)v;
    }
  }
#pragma clang loop unroll(disable)
  for (int j = 0; j < (int)(sizeof(g_vtab) / sizeof(g_vtab[0])); ++j) {
    const VDesc d = g_vtab[j];
    const float* src = P.p[d.idx];
#pragma clang loop unroll(disable)
    for (int i = tid; i < d.n; i += 128) V[d.off + i] = src[i];
  }
  // folded BatchNorm: s = gamma*rsqrt(rv+eps), t = beta - rm*s
  {
    const int bnoff[4][3] = {{VF_AS, VF_AT, 64}, {VF_BS, VF_BT, 32},
                             {VF_CS, VF_CT, 64}, {VF_FS, VF_FT, 64}};
    const short bnidx[4] = {A1_BETA, B1_BETA, C1_BETA, F1_BETA};
#pragma clang loop unroll(disable)
    for (int j = 0; j < 4; ++j) {
      const float* beta  = P.p[bnidx[j] + 0];
      const float* gamma = P.p[bnidx[j] + 1];
      const float* rm    = P.p[bnidx[j] + 2];
      const float* rv    = P.p[bnidx[j] + 3];
#pragma clang loop unroll(disable)
      for (int i = tid; i < bnoff[j][2]; i += 128) {
        float s = gamma[i] * rsqrtf(rv[i] + 1e-5f);
        V[bnoff[j][0] + i] = s;
        V[bnoff[j][1] + i] = beta[i] - rm[i] * s;
      }
    }
  }
  __syncthreads();

  // ---- per-wave staging ----------------------------------------------------
  char* wb = smem + WAVE_OFF + wv * WAVE_BYTES;
  _Float16* act0 = (_Float16*)wb + WO_ACT0_H;
  _Float16* act1 = (_Float16*)wb + WO_ACT1_H;
  _Float16* gc   = (_Float16*)wb + WO_GC_H;
  _Float16* ply  = (_Float16*)wb + WO_PLY_H;   // [16][10][16]
  _Float16* qh   = (_Float16*)wb + WO_QKV_H;   // [80][16], row = m*5+p5
  _Float16* kh   = qh + 1280;
  _Float16* vh   = kh + 1280;
  float* scw = (float*)(wb + WO_SCW_B);
  float* sc3 = (float*)(wb + WO_SC3_B);

  const int s0 = (blockIdx.x * 4 + wv) * 16;
  const int nl = lane & 15;

  // ---- head A: 20 -> 64 (LReLU,BN) -> 32 (LReLU) -> 1 ---------------------
  stage_in(P.p[IN_X1] + (long)s0 * 20, 20, 20, act0, lane);
  dense(act0, 32, WH + WH_A0, V + VF_A0B, V + VF_AS, V + VF_AT, true,  act1, 64, 64, 64, lane);
  dense(act1, 64, WH + WH_A2, V + VF_A2B, nullptr, nullptr,     true,  act0, 32, 32, 32, lane);
  if (lane < 16) {
    float acc = V[VF_A3B];
    for (int k = 0; k < 32; ++k) acc += (float)act0[lane * 32 + k] * V[VF_A3W + k];
    sc3[lane * 3 + 0] = acc;
  }
  WAIT_DS();

  // ---- head B: 10 -> 32 (LReLU,BN) -> 16 (LReLU) -> 1 ---------------------
  stage_in(P.p[IN_X2] + (long)s0 * 10, 10, 10, act0, lane);
  dense(act0, 32, WH + WH_B0, V + VF_B0B, V + VF_BS, V + VF_BT, true,  act1, 32, 32, 32, lane);
  dense(act1, 32, WH + WH_B2, V + VF_B2B, nullptr, nullptr,     true,  act0, 16, 16, 16, lane);
  if (lane < 16) {
    float acc = V[VF_B3B];
    for (int k = 0; k < 16; ++k) acc += (float)act0[lane * 16 + k] * V[VF_B3W + k];
    sc3[lane * 3 + 1] = acc;
  }
  WAIT_DS();

  // ---- per-player G nets + F head -> ply[16][10][16] ----------------------
  for (int p = 0; p < 10; ++p) {
    stage_in(P.p[IN_G1] + (long)(s0 * 10 + p) * 12, 120, 12, act0, lane);
    dense(act0, 32, WH + WH_G10, V + VF_G10B, nullptr, nullptr, true,  act1, 64, 64, 64, lane);
    dense(act1, 64, WH + WH_G11, V + VF_G11B, nullptr, nullptr, true,  act0, 32, 32, 32, lane);
    dense(act0, 32, WH + WH_G12, V + VF_G12B, nullptr, nullptr, false, gc + 0, 64, 8, 16, lane);

    stage_in(P.p[IN_G2] + (long)(s0 * 10 + p) * 8, 80, 8, act0, lane);
    dense(act0, 32, WH + WH_G20, V + VF_G20B, nullptr, nullptr, true,  act1, 32, 32, 32, lane);
    dense(act1, 32, WH + WH_G21, V + VF_G21B, nullptr, nullptr, false, gc + 8, 64, 8, 16, lane);

    stage_in(P.p[IN_G3] + (long)(s0 * 10 + p) * 6, 60, 6, act0, lane);
    dense(act0, 32, WH + WH_G30, V + VF_G30B, nullptr, nullptr, true,  act1, 32, 32, 32, lane);
    dense(act1, 32, WH + WH_G31, V + VF_G31B, nullptr, nullptr, false, gc + 16, 64, 8, 16, lane);

    stage_in(P.p[IN_G4] + (long)(s0 * 10 + p) * 6, 60, 6, act0, lane);
    dense(act0, 32, WH + WH_G40, V + VF_G40B, nullptr, nullptr, true, act1, 16, 16, 16, lane);
    dense_k16(act1, nl * 16, WH + WH_G41, V + VF_G41B, false, gc + 24, 64, 8, 16, lane);

    stage_in(P.p[IN_G5] + (long)(s0 * 10 + p) * 4, 40, 4, act0, lane);
    dense(act0, 32, WH + WH_G50, V + VF_G50B, nullptr, nullptr, true, act1, 16, 16, 16, lane);
    dense_k16(act1, nl * 16, WH + WH_G51, V + VF_G51B, false, gc + 32, 64, 8, 16, lane);

    for (int i = lane; i < 16 * 16; i += 32) {        // zero gc cols 48..63
      int m = i >> 4, k2 = 48 + (i & 15);
      gc[m * 64 + k2] = (_Float16)0.f;
    }
    WAIT_DS();

    // F head3: 40 -> 64 (LReLU,BN) -> 32 (LReLU) -> 16
    dense(gc,   64, WH + WH_F0, V + VF_F0B, V + VF_FS, V + VF_FT, true,  act1, 64, 64, 64, lane);
    dense(act1, 64, WH + WH_F2, V + VF_F2B, nullptr, nullptr,     true,  act0, 32, 32, 32, lane);
    dense(act0, 32, WH + WH_F3, V + VF_F3B, nullptr, nullptr,     false, ply + p * 16, 160, 16, 16, lane);
  }

  // ---- attention pool per team -> gc[16][32] = concat(t1,t2) --------------
  for (int t = 0; t < 2; ++t) {
    const _Float16* wq  = WH + (t ? WH_EQ  : WH_DQ);
    const _Float16* wk  = WH + (t ? WH_EK  : WH_DK);
    const _Float16* wvv = WH + (t ? WH_EV  : WH_DV);
    const _Float16* wo0 = WH + (t ? WH_EO0 : WH_DO0);
    const _Float16* wo1 = WH + (t ? WH_EO1 : WH_DO1);
    const float* bq  = V + (t ? VF_EQB  : VF_DQB);
    const float* bk  = V + (t ? VF_EKB  : VF_DKB);
    const float* bv  = V + (t ? VF_EVB  : VF_DVB);
    const float* bo0 = V + (t ? VF_EO0B : VF_DO0B);
    const float* bo1 = V + (t ? VF_EO1B : VF_DO1B);

    // Q/K/V: 80 team rows (r = m*5+p5) -> 5 WMMA tiles, A-frags straight
    // from ply via per-lane row remap (no staging copies).
    for (int i = 0; i < 5; ++i) {
      int r = i * 16 + nl;
      int m = r / 5, p5 = r - m * 5;
      int rowOff = m * 160 + (t * 5 + p5) * 16;
      dense_k16(ply, rowOff, wq,  bq, false, qh + i * 256, 16, 16, 16, lane);
      dense_k16(ply, rowOff, wk,  bk, false, kh + i * 256, 16, 16, 16, lane);
      dense_k16(ply, rowOff, wvv, bv, false, vh + i * 256, 16, 16, 16, lane);
    }
    // scores (QK^T)/4, softmax over k
    for (int task = lane; task < 80; task += 32) {
      int m = task / 5, q = task % 5;
      float sv[5], mx = -1e30f;
      for (int k2 = 0; k2 < 5; ++k2) {
        float s = 0.f;
        for (int d = 0; d < 16; ++d)
          s += (float)qh[(m * 5 + q) * 16 + d] * (float)kh[(m * 5 + k2) * 16 + d];
        s *= 0.25f;
        sv[k2] = s;
        if (s > mx) mx = s;
      }
      float den = 0.f;
      for (int k2 = 0; k2 < 5; ++k2) { float e = __expf(sv[k2] - mx); sv[k2] = e; den += e; }
      float inv = 1.f / den;
      for (int k2 = 0; k2 < 5; ++k2) scw[(m * 5 + q) * 5 + k2] = sv[k2] * inv;
    }
    WAIT_DS();
    // pooled = mean_q(w @ V) -> act0[16][16]
    for (int i = lane; i < 16 * 16; i += 32) {
      int m = i >> 4, d = i & 15;
      float acc = 0.f;
      for (int q = 0; q < 5; ++q)
        for (int k2 = 0; k2 < 5; ++k2)
          acc += scw[(m * 5 + q) * 5 + k2] * (float)vh[(m * 5 + k2) * 16 + d];
      act0[i] = (_Float16)(acc * 0.2f);
    }
    WAIT_DS();
    dense_k16(act0, nl * 16, wo0, bo0, true, act1, 32, 32, 32, lane);
    dense(act1, 32, wo1, bo1, nullptr, nullptr, false, gc + t * 16, 32, 16, 16, lane);
  }

  // ---- head C: 32 -> 64 (LReLU,BN) -> 32 (LReLU) -> 1 ---------------------
  dense(gc,   32, WH + WH_C0, V + VF_C0B, V + VF_CS, V + VF_CT, true, act1, 64, 64, 64, lane);
  dense(act1, 64, WH + WH_C2, V + VF_C2B, nullptr, nullptr,     true, act0, 32, 32, 32, lane);
  if (lane < 16) {
    float acc = V[VF_C3B];
    for (int k = 0; k < 32; ++k) acc += (float)act0[lane * 32 + k] * V[VF_C3W + k];
    sc3[lane * 3 + 2] = acc;
  }
  WAIT_DS();

  // ---- final: 3 -> 16 (LReLU) -> 1 ----------------------------------------
  if (lane < 16) {
    int m = lane;
    float a0 = sc3[m * 3 + 0], a1 = sc3[m * 3 + 1], a2 = sc3[m * 3 + 2];
    float y = V[VF_FIN1B];
    for (int o = 0; o < 16; ++o) {
      float h = V[VF_FIN0B + o] + V[VF_FIN0W + o * 3 + 0] * a0 +
                V[VF_FIN0W + o * 3 + 1] * a1 + V[VF_FIN0W + o * 3 + 2] * a2;
      h = (h > 0.f) ? h : 0.2f * h;
      y += h * V[VF_FIN1W + o];
    }
    out[s0 + m] = y;
  }
}

// ---------------------------------------------------------------------------
extern "C" void kernel_launch(void* const* d_in, const int* in_sizes, int n_in,
                              void* d_out, int out_size, void* d_ws, size_t ws_size,
                              hipStream_t stream) {
  (void)in_sizes; (void)d_ws; (void)ws_size; (void)out_size;
  P93 P;
  for (int i = 0; i < N_PTRS; ++i)
    P.p[i] = (const float*)d_in[(i < n_in) ? i : 0];
  const int B = 65536;
  dim3 grid(B / 64), block(128);  // 4 waves/block, 16 samples/wave
  hipLaunchKernelGGL(gold_kernel, grid, block, SMEM_BYTES, stream, P, (float*)d_out);
}
